// CircularSplineLayer_72181220376724
// MI455X (gfx1250) — compile-verified, matrix operand order checked by reading, not run on hardware
//
#include <hip/hip_runtime.h>
#include <hip/hip_bf16.h>
#include <math.h>

#define SIZE_HALF 256
#define NSEG      8
#define HIDDEN    512
#define BATCH     8192
#define NIN       512      // 2*SIZE_HALF
#define NOUT      6144     // SIZE_HALF*3*NSEG
#define TWO_PI_F  6.283185307179586f
#define EPS_F     1e-6f

typedef __attribute__((ext_vector_type(16))) __bf16 v16bf;
typedef __attribute__((ext_vector_type(8)))  float  v8f;

static __device__ __forceinline__ unsigned short f2bf(float f) {
  unsigned u = __float_as_uint(f);
  unsigned r = 0x7FFFu + ((u >> 16) & 1u);
  return (unsigned short)((u + r) >> 16);
}

// LDS byte offset of a __shared__-derived pointer (flat addr low 32 bits == LDS offset)
static __device__ __forceinline__ unsigned lds_off(const void* p) {
  return (unsigned)(unsigned long long)(uintptr_t)p;
}

// CDNA5 async DMA: global -> LDS, 16 bytes per lane, tracked by ASYNCcnt
static __device__ __forceinline__ void async_b128(unsigned ldsoff, const void* g) {
  asm volatile("global_load_async_to_lds_b128 %0, %1, off"
               :: "v"(ldsoff), "v"(g) : "memory");
}
static __device__ __forceinline__ void wait_async0() {
  asm volatile("s_wait_asynccnt 0x0" ::: "memory");
}

// ---------------- prep kernels ----------------

// Transpose + convert: W (512 x Ncols, f32, K-major) -> Wt (Ncols x 512, bf16, N-major)
__launch_bounds__(256)
__global__ void k_cvt_t(const float* __restrict__ W, unsigned short* __restrict__ Wt,
                        int Ncols) {
  __shared__ float tile[32][33];
  const int nT = blockIdx.x * 32;    // column tile (N)
  const int kT = blockIdx.y * 32;    // row tile (K)
  const int c  = threadIdx.x & 31;
  const int r0 = threadIdx.x >> 5;   // 0..7
  #pragma unroll
  for (int j = 0; j < 4; ++j) {
    int r = r0 + j * 8;
    tile[r][c] = W[(size_t)(kT + r) * Ncols + nT + c];   // coalesced read
  }
  __syncthreads();
  #pragma unroll
  for (int j = 0; j < 4; ++j) {
    int n = r0 + j * 8;
    Wt[(size_t)(nT + n) * HIDDEN + kT + c] = f2bf(tile[c][n]);  // coalesced write
  }
}

// builds x_for_net (bf16), copies x_a into phi_out[:, :256], inits log_out
__global__ void k_prep(const float* __restrict__ x,
                       const float* __restrict__ logd,
                       unsigned short* __restrict__ xnet,
                       float* __restrict__ phi_out,
                       float* __restrict__ log_out) {
  int i = blockIdx.x * blockDim.x + threadIdx.x;
  if (i >= BATCH * NIN) return;
  int b = i >> 9;
  int j = i & 511;
  float v;
  if (j < SIZE_HALF) {
    float xa = x[i];
    v = __cosf(xa);
    phi_out[i] = xa;                 // phi_out[:, :256] = x_a
  } else {
    float xa = x[b * NIN + (j - SIZE_HALF)];
    v = __sinf(xa);
  }
  xnet[i] = f2bf(v);
  if (j == 0) log_out[b] = logd[b];
}

// ---------------- GEMM1: h = tanh(Xn @ W1 + b1), bf16 out ----------------
// block: 256 thr (8 waves). tile M=128, N=32, K-step 32, async double-buffered.

__launch_bounds__(256)
__global__ void k_gemm1(const unsigned short* __restrict__ Xn,    // 8192x512 bf16
                        const unsigned short* __restrict__ W1t,   // 512x512  bf16 (N x K)
                        const float* __restrict__ b1,             // 512
                        unsigned short* __restrict__ Hb) {        // 8192x512 bf16
  __shared__ __align__(64) unsigned short Abuf[2][128 * 32];  // 2 x 8KB
  __shared__ __align__(64) unsigned short Bbuf[2][32 * 32];   // 2 x 2KB, [col*32 + k]
  const int tid  = threadIdx.x;
  const int wave = tid >> 5;
  const int lane = tid & 31;
  const int rowTile = blockIdx.x * 128;
  const int nBase   = blockIdx.y * 32;

  // straight-line staging: per-thread chunk addresses fixed, only kb shifts
  const unsigned short* gA0 = &Xn[(size_t)(rowTile + (tid >> 2)) * NIN + (tid & 3) * 8];
  const unsigned short* gA1 = gA0 + (size_t)64 * NIN;                 // chunk tid+256
  const unsigned short* gB0 = &W1t[(size_t)(nBase + (tid >> 2)) * HIDDEN + (tid & 3) * 8];
  const unsigned aoff0 = lds_off(&Abuf[0][0]) + tid * 16u;
  const unsigned aoff1 = lds_off(&Abuf[1][0]) + tid * 16u;
  const unsigned boff0 = lds_off(&Bbuf[0][0]) + tid * 16u;
  const unsigned boff1 = lds_off(&Bbuf[1][0]) + tid * 16u;

  #define STAGE1(AOFF, BOFF, KB)                                  \
    do {                                                          \
      async_b128((AOFF),        gA0 + (KB));                      \
      async_b128((AOFF) + 4096, gA1 + (KB));                      \
      if (tid < 128) async_b128((BOFF), gB0 + (KB));              \
    } while (0)

  v8f c0 = {}; v8f c1 = {};

  STAGE1(aoff0, boff0, 0);
  for (int kb = 0; kb < HIDDEN; kb += 32) {
    const int cur = (kb >> 5) & 1;
    wait_async0();
    __syncthreads();
    if (kb + 32 < HIDDEN) {
      if (cur) STAGE1(aoff0, boff0, kb + 32); else STAGE1(aoff1, boff1, kb + 32);
    }
    const int half = lane >> 4, l15 = lane & 15;
    v16bf a  = *(const v16bf*)(&Abuf[cur][(wave * 16 + l15) * 32 + half * 16]);
    v16bf bA = *(const v16bf*)(&Bbuf[cur][l15 * 32 + half * 16]);
    v16bf bB = *(const v16bf*)(&Bbuf[cur][(16 + l15) * 32 + half * 16]);
    c0 = __builtin_amdgcn_wmma_f32_16x16x32_bf16(false, a, false, bA, (short)0, c0, false, false);
    c1 = __builtin_amdgcn_wmma_f32_16x16x32_bf16(false, a, false, bB, (short)0, c1, false, false);
  }
  #undef STAGE1

  const int half = lane >> 4, l15 = lane & 15;
  #pragma unroll
  for (int i = 0; i < 8; ++i) {
    int r = rowTile + wave * 16 + i + half * 8;
    int cc = nBase + l15;
    Hb[(size_t)r * HIDDEN + cc]      = f2bf(tanhf(c0[i] + b1[cc]));
    Hb[(size_t)r * HIDDEN + cc + 16] = f2bf(tanhf(c1[i] + b1[cc + 16]));
  }
}

// ------- GEMM2 fused with spline: tile M=128, N=96 (= 4 spline sites) -------

__launch_bounds__(256)
__global__ void k_gemm2_spline(const unsigned short* __restrict__ Hb,   // 8192x512 bf16
                               const unsigned short* __restrict__ W2t,  // 6144x512 bf16 (N x K)
                               const float* __restrict__ b2,            // 6144
                               const float* __restrict__ x_input,       // 8192x512 f32
                               const float* __restrict__ phase_shift,   // 1
                               float* __restrict__ phi_out,             // 8192x512
                               float* __restrict__ log_out) {           // 8192
  // A/B double buffers (28KB) alias the 48KB C-spill region (used strictly later)
  __shared__ __align__(64) unsigned char smem[128 * 96 * 4 + 128 * 4];
  unsigned short* A0 = (unsigned short*)smem;            // 8KB
  unsigned short* A1 = A0 + 128 * 32;                    // 8KB
  unsigned short* B0 = A1 + 128 * 32;                    // 6KB  [col*32 + k]
  unsigned short* B1 = B0 + 96 * 32;                     // 6KB
  float* Cls    = (float*)smem;                          // 48KB (after K-loop)
  float* rowLog = (float*)(smem + 128 * 96 * 4);         // 512B

  const int tid  = threadIdx.x;
  const int wave = tid >> 5;
  const int lane = tid & 31;
  const int rowTile = blockIdx.x * 128;
  const int colBase = blockIdx.y * 96;

  // prefetch epilogue operands (consumed only after the K-loop)
  __builtin_prefetch(&x_input[(size_t)(rowTile + (tid >> 1)) * NIN + SIZE_HALF + blockIdx.y * 4], 0, 0);
  if (tid == 0) __builtin_prefetch(&b2[colBase], 0, 0);

  // straight-line staging addresses
  const unsigned short* gA0 = &Hb[(size_t)(rowTile + (tid >> 2)) * HIDDEN + (tid & 3) * 8];
  const unsigned short* gA1 = gA0 + (size_t)64 * HIDDEN;              // chunk tid+256
  const unsigned short* gB0 = &W2t[(size_t)(colBase + (tid >> 2)) * HIDDEN + (tid & 3) * 8];
  const unsigned short* gB1 = gB0 + (size_t)64 * HIDDEN;              // cols 64..95, tid<128
  const unsigned aoff0 = lds_off(A0) + tid * 16u;
  const unsigned aoff1 = lds_off(A1) + tid * 16u;
  const unsigned boff0 = lds_off(B0) + tid * 16u;
  const unsigned boff1 = lds_off(B1) + tid * 16u;

  #define STAGE2(AOFF, BOFF, KB)                                  \
    do {                                                          \
      async_b128((AOFF),        gA0 + (KB));                      \
      async_b128((AOFF) + 4096, gA1 + (KB));                      \
      async_b128((BOFF),        gB0 + (KB));                      \
      if (tid < 128) async_b128((BOFF) + 4096, gB1 + (KB));       \
    } while (0)

  v8f c[6] = {};

  STAGE2(aoff0, boff0, 0);
  for (int kb = 0; kb < HIDDEN; kb += 32) {
    const int cur = (kb >> 5) & 1;
    const unsigned short* Ac = cur ? A1 : A0;
    const unsigned short* Bc = cur ? B1 : B0;
    wait_async0();
    __syncthreads();
    if (kb + 32 < HIDDEN) {
      if (cur) STAGE2(aoff0, boff0, kb + 32); else STAGE2(aoff1, boff1, kb + 32);
    }
    const int half = lane >> 4, l15 = lane & 15;
    v16bf a = *(const v16bf*)(&Ac[(wave * 16 + l15) * 32 + half * 16]);
    #pragma unroll
    for (int f = 0; f < 6; ++f) {
      v16bf b = *(const v16bf*)(&Bc[(f * 16 + l15) * 32 + half * 16]);
      c[f] = __builtin_amdgcn_wmma_f32_16x16x32_bf16(false, a, false, b, (short)0, c[f], false, false);
    }
  }
  #undef STAGE2
  __syncthreads();   // all waves done with A/B LDS before aliasing it with Cls

  // spill C fragments to LDS (net_out tile never touches HBM)
  {
    const int half = lane >> 4, l15 = lane & 15;
    #pragma unroll
    for (int f = 0; f < 6; ++f)
      #pragma unroll
      for (int i = 0; i < 8; ++i)
        Cls[(wave * 16 + i + half * 8) * 96 + f * 16 + l15] = c[f][i];
  }
  if (tid < 128) rowLog[tid] = 0.0f;
  __syncthreads();

  const float phase = phase_shift[0];

  // 512 spline items = 128 rows x 4 spline sites
  for (int item = tid; item < 512; item += 256) {
    int r  = item >> 2;                  // 0..127
    int sl = item & 3;                   // 0..3
    int rowG = rowTile + r;
    int sG   = blockIdx.y * 4 + sl;
    const float* cp = &Cls[r * 96 + sl * 24];
    const float* bp = &b2[colBase + sl * 24];

    float hr[8], wr[8], dr[8];
    #pragma unroll
    for (int q = 0; q < 8; ++q) {
      hr[q] = cp[q]      + bp[q];
      wr[q] = cp[q + 8]  + bp[q + 8];
      dr[q] = cp[q + 16] + bp[q + 16];
    }
    float hm = hr[0], wm = wr[0];
    #pragma unroll
    for (int q = 1; q < 8; ++q) { hm = fmaxf(hm, hr[q]); wm = fmaxf(wm, wr[q]); }
    float hn[8], wn[8], dn[8];
    float hs = 0.f, wsum = 0.f;
    #pragma unroll
    for (int q = 0; q < 8; ++q) {
      hn[q] = __expf(hr[q] - hm); hs += hn[q];
      wn[q] = __expf(wr[q] - wm); wsum += wn[q];
    }
    float hiv = TWO_PI_F / hs, wiv = TWO_PI_F / wsum;
    #pragma unroll
    for (int q = 0; q < 8; ++q) {
      hn[q] *= hiv; wn[q] *= wiv;
      float d = dr[q];
      dn[q] = (d > 20.f) ? d : log1pf(__expf(d));   // softplus
    }
    // knots: x_knots = [-EPS, cumsum(w)], phi_knots = [0, cumsum(h)]
    float xk[9], pk[9];
    xk[0] = -EPS_F; pk[0] = 0.f;
    float cw = 0.f, ch = 0.f;
    #pragma unroll
    for (int q = 0; q < 8; ++q) { cw += wn[q]; ch += hn[q]; xk[q + 1] = cw; pk[q + 1] = ch; }

    float xb = x_input[(size_t)rowG * NIN + SIZE_HALF + sG];
    int k = -1;                           // searchsorted(left) - 1
    #pragma unroll
    for (int q = 0; q < 9; ++q) k += (xk[q] < xb) ? 1 : 0;
    k = (k < 0) ? 0 : ((k > 7) ? 7 : k);

    float wk = wn[k], hk = hn[k], sk = hk / wk;
    float dk = dn[k], dk1 = dn[(k + 1) & 7];
    float al  = (xb - xk[k]) / wk;
    float a1m = al * (1.f - al);
    float den = sk + (dk1 + dk - 2.f * sk) * a1m;
    float phi = pk[k] + hk * (sk * al * al + dk * a1m) / den;
    phi += phase;
    phi -= TWO_PI_F * floorf(phi / TWO_PI_F);
    float om = 1.f - al;
    float grad = sk * sk * (dk1 * al * al + 2.f * sk * a1m + dk * om * om) / (den * den);

    phi_out[(size_t)rowG * NIN + SIZE_HALF + sG] = phi;
    atomicAdd(&rowLog[r], __logf(grad));
  }
  __syncthreads();
  if (tid < 128) atomicAdd(&log_out[rowTile + tid], -rowLog[tid]);
}

// ---------------- launcher ----------------

extern "C" void kernel_launch(void* const* d_in, const int* in_sizes, int n_in,
                              void* d_out, int out_size, void* d_ws, size_t ws_size,
                              hipStream_t stream) {
  const float* x_input     = (const float*)d_in[0];
  const float* log_density = (const float*)d_in[1];
  const float* W1          = (const float*)d_in[2];
  const float* b1          = (const float*)d_in[3];
  const float* W2          = (const float*)d_in[4];
  const float* b2          = (const float*)d_in[5];
  const float* phase       = (const float*)d_in[6];
  (void)in_sizes; (void)n_in; (void)out_size; (void)ws_size;

  float* phi_out = (float*)d_out;                       // 8192 x 512
  float* log_out = phi_out + (size_t)BATCH * NIN;       // 8192

  unsigned char* ws = (unsigned char*)d_ws;
  unsigned short* W1t = (unsigned short*)(ws);                        // 512 KB (N x K)
  unsigned short* W2t = (unsigned short*)(ws + 524288);               // 6 MB  (N x K)
  unsigned short* Xn  = (unsigned short*)(ws + 524288 + 6291456);     // 8 MB
  unsigned short* Hb  = (unsigned short*)(ws + 524288 + 6291456 + 8388608); // 8 MB

  dim3 gt1(NIN / 32, HIDDEN / 32);    // 16 x 16
  k_cvt_t<<<gt1, 256, 0, stream>>>(W1, W1t, NIN);
  dim3 gt2(NOUT / 32, HIDDEN / 32);   // 192 x 16
  k_cvt_t<<<gt2, 256, 0, stream>>>(W2, W2t, NOUT);

  k_prep<<<(BATCH * NIN) / 256, 256, 0, stream>>>(x_input, log_density, Xn, phi_out, log_out);

  dim3 g1(BATCH / 128, NIN / 32);    // 64 x 16
  k_gemm1<<<g1, 256, 0, stream>>>(Xn, W1t, b1, Hb);

  dim3 g2(BATCH / 128, NOUT / 96);   // 64 x 64
  k_gemm2_spline<<<g2, 256, 0, stream>>>(Hb, W2t, b2, x_input, phase, phi_out, log_out);
}
